// SparseSelfAttention_28174985462025
// MI455X (gfx1250) — compile-verified, matrix operand order checked by reading, not run on hardware
//
#include <hip/hip_runtime.h>
#include <hip/hip_bf16.h>

// ---------------------------------------------------------------------------
// SparseSelfAttention on MI455X (gfx1250, wave32, WMMA bf16 path).
// All operand layouts arranged K-contiguous so every WMMA fragment is built
// from dword LDS loads (no u16 gathers anywhere):
//   Xt[b]   = x[b]^T                (4096x512  bf16, fused transpose-convert)
//   QK[b]   = Wqk(2048x512) @ Xt^T  (2048x4096 bf16)   A=Wqk,  B=Xt   (TRANS)
//   Vt[b]   = Xt(4096x512) @ Wv^T   (4096x1024 bf16)   A=Xt,   B=Wv   (TRANS)
//   att     = q @ k^T per (b,h)     (128x128   f32)    A=q,    B=k    (TRANS)
//   att     = softmax(att) * (mask<0.1)        -> bf16
//   AOt     = Vt @ att^T per (b,h)  (4096x128  bf16)   A=Vt,   B=att  (TRANS)
//   out     = Wout @ AOt^T + b_out  (512x4096  f32)    A=Wout, B=AOt  (TRANS)
// ---------------------------------------------------------------------------

#define BATCH   8
#define CHAN    512
#define NPIX    4096          // 64*64
#define NHEADS  8
#define HDIM    128
#define EDIM    1024          // NHEADS*HDIM
#define E3      3072          // 3*EDIM
#define E2      2048          // q+k rows

typedef __attribute__((ext_vector_type(16))) __bf16         bf16x16;
typedef __attribute__((ext_vector_type(8)))  float          floatx8;
typedef __attribute__((ext_vector_type(8)))  unsigned int   uintx8;
typedef __attribute__((ext_vector_type(4)))  unsigned int   uintx4;

__device__ __forceinline__ unsigned short f2bf(float f) {
    unsigned u = __builtin_bit_cast(unsigned, f);
    unsigned r = u + 0x7FFFu + ((u >> 16) & 1u);   // round-to-nearest-even
    return (unsigned short)(r >> 16);
}

// --------------------------- f32 -> bf16 convert ---------------------------
__global__ void __launch_bounds__(256)
ssa_f2bf_kernel(const float* __restrict__ s, unsigned short* __restrict__ d,
                long long n) {
    long long i = (long long)blockIdx.x * 256 + threadIdx.x;
    long long stride = (long long)gridDim.x * 256;
    for (; i < n; i += stride) d[i] = f2bf(s[i]);
}

// ----------------- fused transpose + convert: x[b] -> Xt[b] ----------------
// src (CHAN x NPIX) f32 row-major -> dst (NPIX x CHAN) bf16 row-major.
__global__ void __launch_bounds__(256)
ssa_xpose_f2bf(const float* __restrict__ src, unsigned short* __restrict__ dst) {
    const float*    s = src + (size_t)blockIdx.z * CHAN * NPIX;
    unsigned short* d = dst + (size_t)blockIdx.z * NPIX * CHAN;
    const int n0 = blockIdx.x * 32;
    const int c0 = blockIdx.y * 32;
    __shared__ float t[32][33];
    const int tx = threadIdx.x & 31, ty = threadIdx.x >> 5;
#pragma unroll
    for (int i = 0; i < 4; ++i)
        t[ty + i * 8][tx] = s[(size_t)(c0 + ty + i * 8) * NPIX + n0 + tx];
    __syncthreads();
#pragma unroll
    for (int i = 0; i < 4; ++i)
        d[(size_t)(n0 + ty + i * 8) * CHAN + c0 + tx] = f2bf(t[tx][ty + i * 8]);
}

// --------------------------- tiled bf16 WMMA GEMM --------------------------
// C[z](M x N) = A[z](M x K) * B[z]^T (+ bias), B given as (N x K) row-major.
// z -> (zo=z/ZIN, zi=z%ZIN); ptr += zo*S?O + zi*S?I (flat element offsets).
// Block: 256 threads = 8 waves (4 in M x 2 in N). Tile 128x128x64.
// Wave tile 32(M) x 64(N): 2x4 v8f accumulators, 16 WMMAs per K-stage.
// BIAS: 0 = none, 1 = per-M-row, 2 = per-N-col.
template<int Mt, int Nt, int Kt, int LDA, int LDB, int LDC, int ZIN,
         long long SAO, long long SAI, long long SBO, long long SBI,
         long long SCO, long long SCI, int BIAS, bool OUT_BF16>
__global__ void __launch_bounds__(256)
ssa_gemm_wmma(const unsigned short* __restrict__ Aall,
              const unsigned short* __restrict__ Ball,
              const float* __restrict__ bias,
              void* __restrict__ Call) {
    const int z  = blockIdx.z;
    const int zo = z / ZIN;
    const int zi = z % ZIN;
    const unsigned short* A  = Aall + zo * SAO + zi * SAI;
    const unsigned short* Bm = Ball + zo * SBO + zi * SBI;

    const int m0 = blockIdx.y * 128;
    const int n0 = blockIdx.x * 128;

    // Row stride 72 ushorts (144 B): rows 16B-aligned -> b128 LDS stores,
    // dword-aligned fragment loads.
    __shared__ unsigned short As[128][72];   // [m][k]
    __shared__ unsigned short Bs[128][72];   // [n][k]

    const int tid  = threadIdx.x;
    const int lane = tid & 31;
    const int wave = tid >> 5;
    const int half = (lane >> 4) & 1;        // wave32 lo/hi half (ISA 7.12.2)
    const int l16  = lane & 15;
    const int wm   = (wave >> 1) * 32;       // wave M origin within tile
    const int wn   = (wave & 1) * 64;        // wave N origin within tile

    floatx8 zero = {0.f, 0.f, 0.f, 0.f, 0.f, 0.f, 0.f, 0.f};
    floatx8 acc[2][4];
#pragma unroll
    for (int i = 0; i < 2; ++i)
#pragma unroll
        for (int j = 0; j < 4; ++j) acc[i][j] = zero;

    uintx4 aReg[4];   // 128x64 tile: 8192 elems / 256 thr = 4 x b128
    uintx4 bReg[4];

    auto loadA = [&](int k0) {
#pragma unroll
        for (int it = 0; it < 4; ++it) {
            int q = tid + it * 256;
            int r = q >> 3, c = (q & 7) * 8;          // 8 chunks per row
            aReg[it] = *(const uintx4*)(A + (size_t)(m0 + r) * LDA + k0 + c);
        }
    };
    auto loadB = [&](int k0) {
#pragma unroll
        for (int it = 0; it < 4; ++it) {
            int q = tid + it * 256;
            int r = q >> 3, c = (q & 7) * 8;
            bReg[it] = *(const uintx4*)(Bm + (size_t)(n0 + r) * LDB + k0 + c);
        }
    };
    auto storeTiles = [&]() {
#pragma unroll
        for (int it = 0; it < 4; ++it) {
            int q = tid + it * 256;
            int r = q >> 3, c = (q & 7) * 8;
            *(uintx4*)&As[r][c] = aReg[it];
        }
#pragma unroll
        for (int it = 0; it < 4; ++it) {
            int q = tid + it * 256;
            int r = q >> 3, c = (q & 7) * 8;
            *(uintx4*)&Bs[r][c] = bReg[it];
        }
    };

    loadA(0);
    loadB(0);

    for (int k0 = 0; k0 < Kt; k0 += 64) {
        storeTiles();
        __syncthreads();
        if (k0 + 64 < Kt) {       // prefetch next stage while WMMAs run
            loadA(k0 + 64);
            loadB(k0 + 64);
        }

#pragma unroll
        for (int ks = 0; ks < 2; ++ks) {
            // A fragments for both 16-row M tiles of this wave
            bf16x16 afrag[2];
#pragma unroll
            for (int mt = 0; mt < 2; ++mt) {
                uintx8 au;
#pragma unroll
                for (int j = 0; j < 8; ++j) {
                    int kb = ks * 32 + ((j & 4) ? 16 : 0) + (half ? 8 : 0)
                           + ((j & 3) << 1);
                    au[j] = *(const unsigned int*)&As[wm + mt * 16 + l16][kb];
                }
                afrag[mt] = __builtin_bit_cast(bf16x16, au);
            }
#pragma unroll
            for (int nt = 0; nt < 4; ++nt) {
                uintx8 bu;
#pragma unroll
                for (int j = 0; j < 8; ++j) {
                    int kb = ks * 32 + ((j & 4) ? 16 : 0) + (half ? 8 : 0)
                           + ((j & 3) << 1);
                    bu[j] = *(const unsigned int*)&Bs[wn + nt * 16 + l16][kb];
                }
                bf16x16 bfrag = __builtin_bit_cast(bf16x16, bu);
#pragma unroll
                for (int mt = 0; mt < 2; ++mt)
                    acc[mt][nt] = __builtin_amdgcn_wmma_f32_16x16x32_bf16(
                        false, afrag[mt], false, bfrag, (short)0,
                        acc[mt][nt], false, false);
            }
        }
        __syncthreads();
    }

    // ---- epilogue: D layout VGPR r -> row = r + 8*half, col = l16 ----
    const long long coff = zo * SCO + zi * SCI;
#pragma unroll
    for (int mt = 0; mt < 2; ++mt) {
#pragma unroll
        for (int nt = 0; nt < 4; ++nt) {
#pragma unroll
            for (int r = 0; r < 8; ++r) {
                int grow = m0 + wm + mt * 16 + r + half * 8;
                int gcol = n0 + wn + nt * 16 + l16;
                float v = acc[mt][nt][r];
                if (BIAS == 1) v += bias[grow];
                if (BIAS == 2) v += bias[gcol];
                size_t idx = (size_t)coff + (size_t)grow * LDC + gcol;
                if (OUT_BF16)
                    ((unsigned short*)Call)[idx] = f2bf(v);
                else
                    ((float*)Call)[idx] = v;
            }
        }
    }
}

// --------------------- softmax over 128 + sparsity mask --------------------
__global__ void __launch_bounds__(128)
ssa_softmax_mask(const float* __restrict__ att,
                 const float* __restrict__ mask_u,
                 unsigned short* __restrict__ out) {
    const int row = blockIdx.x;        // b*NHEADS*HDIM rows
    const int t   = threadIdx.x;       // 128 = row length
    __shared__ float red[128];
    const size_t base = (size_t)row * HDIM;

    float v = att[base + t];
    red[t] = v;
    __syncthreads();
    for (int s = 64; s > 0; s >>= 1) {
        if (t < s) red[t] = fmaxf(red[t], red[t + s]);
        __syncthreads();
    }
    float m = red[0];
    __syncthreads();
    float e = __expf(v - m);
    red[t] = e;
    __syncthreads();
    for (int s = 64; s > 0; s >>= 1) {
        if (t < s) red[t] += red[t + s];
        __syncthreads();
    }
    float sum = red[0];
    float o = (mask_u[base + t] < 0.1f) ? (e / sum) : 0.0f;
    out[base + t] = f2bf(o);
}

// ---------------------------------------------------------------------------
extern "C" void kernel_launch(void* const* d_in, const int* in_sizes, int n_in,
                              void* d_out, int out_size, void* d_ws, size_t ws_size,
                              hipStream_t stream) {
    const float* x      = (const float*)d_in[0];  // (8,512,64,64)
    const float* w_qkv  = (const float*)d_in[1];  // (3072,512)
    const float* b_qkv  = (const float*)d_in[2];  // (3072)
    const float* w_out  = (const float*)d_in[3];  // (512,1024)
    const float* b_out  = (const float*)d_in[4];  // (512)
    const float* mask_u = (const float*)d_in[5];  // (8,8,128,128)
    float* out = (float*)d_out;                   // (8,512,64,64)

    // ------- workspace carve (all region sizes are multiples of 16 B) ------
    char* ws = (char*)d_ws;
    const long long nXt  = (long long)BATCH * NPIX * CHAN;          // 16.8M
    const long long nWq  = (long long)E3 * CHAN;                    // 1.6M
    const long long nWo  = (long long)CHAN * EDIM;                  // 0.5M
    const long long nQK  = (long long)BATCH * E2 * NPIX;            // 67.1M
    const long long nATT = (long long)BATCH * NHEADS * HDIM * HDIM; // 1.05M
    const long long nVT  = (long long)BATCH * NPIX * EDIM;          // 33.6M

    unsigned short* Xt   = (unsigned short*)ws;  ws += nXt  * 2;
    unsigned short* Wqb  = (unsigned short*)ws;  ws += nWq  * 2;
    unsigned short* Wob  = (unsigned short*)ws;  ws += nWo  * 2;
    unsigned short* QKb  = (unsigned short*)ws;  ws += nQK  * 2;
    float*          ATT  = (float*)ws;           ws += nATT * 4;
    unsigned short* ATTb = (unsigned short*)ws;  ws += nATT * 2;
    unsigned short* Vt   = (unsigned short*)ws;  ws += nVT  * 2;
    unsigned short* AOt  = (unsigned short*)ws;  ws += nVT  * 2;

    // ------- 1) conversions: Xt = x^T (bf16); Wq, Wo -> bf16 ---------------
    ssa_xpose_f2bf<<<dim3(NPIX / 32, CHAN / 32, BATCH), 256, 0, stream>>>(x, Xt);
    ssa_f2bf_kernel<<<1024, 256, 0, stream>>>(w_qkv, Wqb, nWq);
    ssa_f2bf_kernel<<<512,  256, 0, stream>>>(w_out, Wob, nWo);

    // ------- 2a) q,k: QK[b](2048x4096) = Wqk(2048x512) @ Xt[b]^T + bias ----
    ssa_gemm_wmma<E2, NPIX, CHAN, CHAN, CHAN, NPIX, 1,
                  0LL, 0LL,
                  (long long)NPIX * CHAN, 0LL,
                  (long long)E2 * NPIX, 0LL,
                  1, true>
        <<<dim3(NPIX / 128, E2 / 128, BATCH), 256, 0, stream>>>(
            Wqb, Xt, b_qkv, QKb);

    // ------- 2b) Vt[b](4096x1024) = Xt[b](4096x512) @ Wv^T + bias(col) -----
    ssa_gemm_wmma<NPIX, EDIM, CHAN, CHAN, CHAN, EDIM, 1,
                  (long long)NPIX * CHAN, 0LL,
                  0LL, 0LL,
                  (long long)NPIX * EDIM, 0LL,
                  2, true>
        <<<dim3(EDIM / 128, NPIX / 128, BATCH), 256, 0, stream>>>(
            Xt, Wqb + (size_t)E2 * CHAN, b_qkv + E2, Vt);

    // ------- 3) att = q @ k^T per (b,h): (128x4096)@(4096x128) -> f32 ------
    ssa_gemm_wmma<HDIM, HDIM, NPIX, NPIX, NPIX, HDIM, NHEADS,
                  (long long)E2 * NPIX, (long long)HDIM * NPIX,
                  (long long)E2 * NPIX, (long long)HDIM * NPIX,
                  (long long)NHEADS * HDIM * HDIM, (long long)HDIM * HDIM,
                  0, false>
        <<<dim3(1, 1, BATCH * NHEADS), 256, 0, stream>>>(
            QKb, QKb + (size_t)EDIM * NPIX, nullptr, ATT);

    // ------- 4) softmax over last dim (128) + sparse mask -> bf16 ----------
    ssa_softmax_mask<<<BATCH * NHEADS * HDIM, 128, 0, stream>>>(ATT, mask_u, ATTb);

    // ------- 5) AOt[b,h](4096x128) = Vt[b,h](4096x128) @ att^T -> bf16 -----
    ssa_gemm_wmma<NPIX, HDIM, HDIM, EDIM, HDIM, EDIM, NHEADS,
                  (long long)NPIX * EDIM, (long long)HDIM,
                  (long long)NHEADS * HDIM * HDIM, (long long)HDIM * HDIM,
                  (long long)NPIX * EDIM, (long long)HDIM,
                  0, true>
        <<<dim3(1, NPIX / 128, BATCH * NHEADS), 256, 0, stream>>>(
            Vt, ATTb, nullptr, AOt);

    // ------- 6) out = Wout(512x1024) @ AOt[b]^T + b_out -> f32 -------------
    ssa_gemm_wmma<CHAN, NPIX, EDIM, EDIM, EDIM, NPIX, 1,
                  0LL, 0LL,
                  (long long)NPIX * EDIM, 0LL,
                  (long long)CHAN * NPIX, 0LL,
                  1, false>
        <<<dim3(NPIX / 128, CHAN / 128, BATCH), 256, 0, stream>>>(
            Wob, AOt, b_out, out);
}
